// GPT_28020366639654
// MI455X (gfx1250) — compile-verified, matrix operand order checked by reading, not run on hardware
//
#include <hip/hip_runtime.h>
#include <hip/hip_bf16.h>
#include <math.h>

// ---------------- model constants ----------------
#define BB   2
#define TT   1024
#define CC   1024
#define HH   16
#define DD   64          // head dim
#define LL   4
#define FFC  4096
#define VV   50257
#define MM   (BB * TT)   // 2048 tokens
#define EPSV 1e-5f

typedef __attribute__((ext_vector_type(16))) _Float16 v16h;
typedef __attribute__((ext_vector_type(8)))  _Float16 v8h;
typedef __attribute__((ext_vector_type(8)))  float    v8f;

// ---------------- GEMM tiling ----------------
// Block tile: 128(M) x 64(N) x 32(K). 8 waves in 4x2; each wave owns a
// 32x32 output macro-tile = 2x2 WMMA accumulators. Double-buffered LDS.
#define BM 128
#define BN 64
#define BK 32

#define FLAG_GELU  1
#define FLAG_ACCUM 2

// ---- CDNA5 async global->LDS copy (16B per lane, ASYNCcnt-tracked) ----
__device__ __forceinline__ void async_cp16(const _Float16* gsrc, _Float16* ldst) {
    // generic pointer to __shared__ keeps the LDS byte offset in the low 32 bits
    unsigned int  loff = (unsigned int)(unsigned long long)(uintptr_t)ldst;
    unsigned long long g = (unsigned long long)(uintptr_t)gsrc;
    asm volatile("global_load_async_to_lds_b128 %0, %1, off"
                 :: "v"(loff), "v"(g) : "memory");
}
__device__ __forceinline__ void wait_async0() {
#if __has_builtin(__builtin_amdgcn_s_wait_asynccnt)
    __builtin_amdgcn_s_wait_asynccnt(0);
#else
    asm volatile("s_wait_asynccnt 0x0" ::: "memory");
#endif
}

// =====================================================================
// Embedding: h[b,t,:] = tok_emb[x[b,t],:] + pos_emb[t,:]
// =====================================================================
__global__ __launch_bounds__(256) void gpt_embed(
    const int* __restrict__ x, const float* __restrict__ tok,
    const float* __restrict__ pos, float* __restrict__ h)
{
    int row = blockIdx.x;            // 0..MM-1
    int t   = row % TT;
    int id  = x[row];
    const float* tr = tok + (size_t)id * CC;
    const float* pr = pos + (size_t)t  * CC;
    float* hr = h + (size_t)row * CC;
    for (int c = threadIdx.x; c < CC; c += 256)
        hr[c] = tr[c] + pr[c];
}

// =====================================================================
// LayerNorm over C, emits f16 activations for the WMMA data path.
// =====================================================================
__global__ __launch_bounds__(256) void gpt_ln_f16(
    const float* __restrict__ xin, const float* __restrict__ g,
    const float* __restrict__ b, _Float16* __restrict__ out)
{
    __shared__ float rs[256], rs2[256];
    int row = blockIdx.x, tid = threadIdx.x;
    const float* xr = xin + (size_t)row * CC;
    float s = 0.f, s2 = 0.f;
    for (int c = tid; c < CC; c += 256) { float v = xr[c]; s += v; s2 += v * v; }
    rs[tid] = s; rs2[tid] = s2;
    __syncthreads();
    for (int st = 128; st > 0; st >>= 1) {
        if (tid < st) { rs[tid] += rs[tid + st]; rs2[tid] += rs2[tid + st]; }
        __syncthreads();
    }
    float mean = rs[0] * (1.0f / CC);
    float var  = rs2[0] * (1.0f / CC) - mean * mean;
    float rstd = rsqrtf(var + EPSV);
    _Float16* orow = out + (size_t)row * CC;
    for (int c = tid; c < CC; c += 256)
        orow[c] = (_Float16)(((xr[c] - mean) * rstd) * g[c] + b[c]);
}

// =====================================================================
// f32 -> f16 elementwise convert
// =====================================================================
__global__ __launch_bounds__(256) void gpt_cvt16(
    const float* __restrict__ in, _Float16* __restrict__ out, int n)
{
    int i = blockIdx.x * 256 + threadIdx.x;
    if (i < n) out[i] = (_Float16)in[i];
}

// =====================================================================
// WMMA GEMM:  Out[M,N] (fp32) = A[M,K] (f16) @ W[N,K]^T (fp32 -> f16)
//
// LDS tiles stored PRE-SWIZZLED into WMMA fragment order
// (tile[t][lane][0..15] = the 16 halfs lane needs for one 16x16x32 op),
// double-buffered. A tile is staged with GLOBAL_LOAD_ASYNC_TO_LDS_B128
// (no VGPR round trip, ASYNCcnt); B tile is software-pipelined through
// VGPRs because it needs the fp32->f16 convert. One barrier per K-step:
//   storeB(cur) ; s_wait_asynccnt 0 ; barrier ;
//   issue asyncA(next) + loadB(next) ; 4x WMMA on cur.
// =====================================================================
__global__ __launch_bounds__(256) void gpt_gemm_f16w32(
    const _Float16* __restrict__ A, const float* __restrict__ W,
    const float* __restrict__ bias, float* __restrict__ Out,
    int M, int N, int K, int flags)
{
    __shared__ _Float16 As2[2][(BM / 16) * 32 * 16];   // 2 x 8 KB
    __shared__ _Float16 Bs2[2][(BN / 16) * 32 * 16];   // 2 x 4 KB

    const int tid  = threadIdx.x;
    const int lane = tid & 31;
    const int wave = tid >> 5;
    const int wy   = wave & 3;   // 4 wave-rows * 32 rows = 128 (M)
    const int wx   = wave >> 2;  // 2 wave-cols * 32 cols = 64  (N)
    const int mBase = blockIdx.y * BM;
    const int nBase = blockIdx.x * BN;

    const int half = lane >> 4;
    const int lr   = lane & 15;

    // per-thread staging geometry (chunk = 8 contiguous halfs of one row)
    const int arow = tid >> 2;          // 0..63 (and +64 for second chunk)
    const int ac   = tid & 3;           // k-offset = ac*8
    // LDS dst for chunk (row, c): lane = (row&15) + (c&1)*16, pos = (c>>1)*8
    const int adl  = (arow & 15) + ((ac & 1) << 4);
    const int adst0 = (((arow >> 4) * 32 + adl) << 4) + ((ac >> 1) << 3);
    const int arow1 = arow + 64;
    const int adl1  = (arow1 & 15) + ((ac & 1) << 4);
    const int adst1 = (((arow1 >> 4) * 32 + adl1) << 4) + ((ac >> 1) << 3);

    const int brow = tid >> 2;          // 0..63
    const int bgn  = nBase + brow;
    const int bdl  = (brow & 15) + ((ac & 1) << 4);
    const int bdst = (((brow >> 4) * 32 + bdl) << 4) + ((ac >> 1) << 3);

    v8f acc00 = {}, acc01 = {}, acc10 = {}, acc11 = {};

    const int nk = K / BK;

    // ---- prologue: async-A tile 0, B tile 0 into registers ----
    async_cp16(A + (size_t)(mBase + arow) * K + ac * 8,        &As2[0][adst0]);
    async_cp16(A + (size_t)(mBase + arow1) * K + ac * 8,       &As2[0][adst1]);
    v8f breg = {};
    if (bgn < N) breg = *(const v8f*)(W + (size_t)bgn * K + ac * 8);

    for (int i = 0; i < nk; ++i) {
        const int buf = i & 1;
        const int k0  = i * BK;

        // ---- convert + store B tile i ----
        {
            v8h hv;
#pragma unroll
            for (int j = 0; j < 8; ++j) hv[j] = (_Float16)breg[j];
            *(v8h*)(&Bs2[buf][bdst]) = hv;
        }

        // ---- A tile i landed? + all waves' stores visible ----
        wait_async0();
        __syncthreads();

        // ---- issue stage for tile i+1 (overlaps with WMMA below) ----
        if (i + 1 < nk) {
            const int kn = k0 + BK;
            async_cp16(A + (size_t)(mBase + arow) * K + kn + ac * 8,  &As2[buf ^ 1][adst0]);
            async_cp16(A + (size_t)(mBase + arow1) * K + kn + ac * 8, &As2[buf ^ 1][adst1]);
            if (bgn < N) breg = *(const v8f*)(W + (size_t)bgn * K + kn + ac * 8);
            if (i + 2 < nk && bgn < N)
                __builtin_prefetch(W + (size_t)bgn * K + kn + BK, 0, 3);
        }

        // ---- fragments: one 32B contiguous LDS read each ----
        const v16h af0 = *(const v16h*)(&As2[buf][(((wy * 2 + 0) * 32) + lane) << 4]);
        const v16h af1 = *(const v16h*)(&As2[buf][(((wy * 2 + 1) * 32) + lane) << 4]);
        const v16h bf0 = *(const v16h*)(&Bs2[buf][(((wx * 2 + 0) * 32) + lane) << 4]);
        const v16h bf1 = *(const v16h*)(&Bs2[buf][(((wx * 2 + 1) * 32) + lane) << 4]);

        acc00 = __builtin_amdgcn_wmma_f32_16x16x32_f16(false, af0, false, bf0,
                                                       (short)0, acc00, false, false);
        acc01 = __builtin_amdgcn_wmma_f32_16x16x32_f16(false, af0, false, bf1,
                                                       (short)0, acc01, false, false);
        acc10 = __builtin_amdgcn_wmma_f32_16x16x32_f16(false, af1, false, bf0,
                                                       (short)0, acc10, false, false);
        acc11 = __builtin_amdgcn_wmma_f32_16x16x32_f16(false, af1, false, bf1,
                                                       (short)0, acc11, false, false);
    }

    // ---- epilogue: C/D layout -> row = 16*tile + half*8 + j, col = lr ----
    v8f accs[2][2] = {{acc00, acc01}, {acc10, acc11}};
#pragma unroll
    for (int s = 0; s < 2; ++s) {
#pragma unroll
        for (int t = 0; t < 2; ++t) {
            const int n = nBase + (wx * 2 + t) * 16 + lr;
            if (n >= N) continue;
            const float bv = bias ? bias[n] : 0.f;
            const int rowBase = mBase + (wy * 2 + s) * 16 + half * 8;
#pragma unroll
            for (int j = 0; j < 8; ++j) {
                float v = accs[s][t][j] + bv;
                if (flags & FLAG_GELU)
                    v = 0.5f * v * (1.0f + erff(v * 0.70710678118f));
                float* p = Out + (size_t)(rowBase + j) * N + n;
                if (flags & FLAG_ACCUM) *p += v;
                else                    *p  = v;
            }
        }
    }
}

// =====================================================================
// Causal attention, one block (128 threads) per (b, head, q).
// =====================================================================
__global__ __launch_bounds__(128) void gpt_attn(
    const float* __restrict__ qkv, float* __restrict__ att)
{
    __shared__ float sc[TT];
    __shared__ float red[128];
    __shared__ float qv[DD];

    const int q  = blockIdx.x;
    const int hh = blockIdx.y;
    const int b  = blockIdx.z;
    const int tid = threadIdx.x;
    const float scale = 0.125f;  // 1/sqrt(64)

    const size_t rowQ = (size_t)(b * TT + q) * (3 * CC);
    if (tid < DD) qv[tid] = qkv[rowQ + hh * DD + tid];
    __syncthreads();

    float lmax = -3.0e38f;
    for (int k = tid; k <= q; k += 128) {
        const float* kr = qkv + (size_t)(b * TT + k) * (3 * CC) + CC + hh * DD;
        float d = 0.f;
#pragma unroll 8
        for (int i = 0; i < DD; ++i) d += qv[i] * kr[i];
        d *= scale;
        sc[k] = d;
        lmax = fmaxf(lmax, d);
    }
    red[tid] = lmax;
    __syncthreads();
    for (int s = 64; s > 0; s >>= 1) {
        if (tid < s) red[tid] = fmaxf(red[tid], red[tid + s]);
        __syncthreads();
    }
    const float m = red[0];
    __syncthreads();

    float lsum = 0.f;
    for (int k = tid; k <= q; k += 128) {
        float e = __expf(sc[k] - m);
        sc[k] = e;
        lsum += e;
    }
    red[tid] = lsum;
    __syncthreads();
    for (int s = 64; s > 0; s >>= 1) {
        if (tid < s) red[tid] += red[tid + s];
        __syncthreads();
    }
    const float inv = 1.0f / red[0];
    __syncthreads();

    if (tid < DD) {
        float acc = 0.f;
        const float* vbase = qkv + (size_t)(b * TT) * (3 * CC) + 2 * CC + hh * DD + tid;
        for (int k = 0; k <= q; ++k)
            acc += sc[k] * vbase[(size_t)k * (3 * CC)];
        att[(size_t)(b * TT + q) * CC + hh * DD + tid] = acc * inv;
    }
}

// =====================================================================
// Host-side orchestration (graph-capture-safe: only launches on stream)
// =====================================================================
extern "C" void kernel_launch(void* const* d_in, const int* in_sizes, int n_in,
                              void* d_out, int out_size, void* d_ws, size_t ws_size,
                              hipStream_t stream) {
    (void)in_sizes; (void)n_in; (void)out_size; (void)ws_size;

    const int*   x       = (const int*)  d_in[0];
    const float* tok_emb = (const float*)d_in[1];
    const float* pos_emb = (const float*)d_in[2];
    const float* qkv_w   = (const float*)d_in[3];   // [L, 3C, C]
    const float* out_w   = (const float*)d_in[4];   // [L, C, C]
    const float* ln1_g   = (const float*)d_in[5];
    const float* ln1_b   = (const float*)d_in[6];
    const float* ln2_g   = (const float*)d_in[7];
    const float* ln2_b   = (const float*)d_in[8];
    const float* fc1_w   = (const float*)d_in[9];   // [L, FF, C]
    const float* fc1_b   = (const float*)d_in[10];  // [L, FF]
    const float* fc2_w   = (const float*)d_in[11];  // [L, C, FF]
    const float* fc2_b   = (const float*)d_in[12];  // [L, C]
    const float* lnf_g   = (const float*)d_in[13];
    const float* lnf_b   = (const float*)d_in[14];
    const float* head_w  = (const float*)d_in[15];  // [V, C]
    float* out = (float*)d_out;

    // workspace layout (bytes)
    char* ws = (char*)d_ws;
    constexpr size_t OFF_H     = 0;                               //  8 MB fp32 residual
    constexpr size_t OFF_ACT16 = OFF_H     + (size_t)MM*CC*4;     // 16 MB f16 activations
    constexpr size_t OFF_QKV   = OFF_ACT16 + (size_t)MM*FFC*2;    // 24 MB fp32 qkv
    constexpr size_t OFF_ATT   = OFF_QKV   + (size_t)MM*3*CC*4;   //  8 MB fp32 attn out
    constexpr size_t OFF_MID   = OFF_ATT   + (size_t)MM*CC*4;     // 32 MB fp32 mlp mid
    float*    h     = (float*)   (ws + OFF_H);
    _Float16* act16 = (_Float16*)(ws + OFF_ACT16);
    float*    qkv   = (float*)   (ws + OFF_QKV);
    float*    att   = (float*)   (ws + OFF_ATT);
    float*    mid   = (float*)   (ws + OFF_MID);

    const dim3 blk256(256), blk128(128);

    gpt_embed<<<MM, blk256, 0, stream>>>(x, tok_emb, pos_emb, h);

    for (int l = 0; l < LL; ++l) {
        // --- attention block ---
        gpt_ln_f16<<<MM, blk256, 0, stream>>>(h, ln1_g + l*CC, ln1_b + l*CC, act16);

        gpt_gemm_f16w32<<<dim3(3*CC/BN, MM/BM), blk256, 0, stream>>>(
            act16, qkv_w + (size_t)l*3*CC*CC, nullptr, qkv, MM, 3*CC, CC, 0);

        gpt_attn<<<dim3(TT, HH, BB), blk128, 0, stream>>>(qkv, att);

        gpt_cvt16<<<(MM*CC + 255)/256, blk256, 0, stream>>>(att, act16, MM*CC);

        gpt_gemm_f16w32<<<dim3(CC/BN, MM/BM), blk256, 0, stream>>>(
            act16, out_w + (size_t)l*CC*CC, nullptr, h, MM, CC, CC, FLAG_ACCUM);

        // --- MLP block ---
        gpt_ln_f16<<<MM, blk256, 0, stream>>>(h, ln2_g + l*CC, ln2_b + l*CC, act16);

        gpt_gemm_f16w32<<<dim3(FFC/BN, MM/BM), blk256, 0, stream>>>(
            act16, fc1_w + (size_t)l*FFC*CC, fc1_b + (size_t)l*FFC, mid,
            MM, FFC, CC, FLAG_GELU);

        gpt_cvt16<<<(MM*FFC + 255)/256, blk256, 0, stream>>>(mid, act16, MM*FFC);

        gpt_gemm_f16w32<<<dim3(CC/BN, MM/BM), blk256, 0, stream>>>(
            act16, fc2_w + (size_t)l*CC*FFC, fc2_b + (size_t)l*CC, h,
            MM, CC, FFC, FLAG_ACCUM);
    }

    // final LN + LM head -> logits [MM, V] fp32
    gpt_ln_f16<<<MM, blk256, 0, stream>>>(h, lnf_g, lnf_b, act16);
    gpt_gemm_f16w32<<<dim3((VV + BN - 1)/BN, MM/BM), blk256, 0, stream>>>(
        act16, head_w, nullptr, out, MM, VV, CC, 0);
}
